// MultiHeadAttention_49014166782041
// MI455X (gfx1250) — matrix operand order repaired and checked by reading, not compiled, not run on hardware
//
#include <hip/hip_runtime.h>
#include <hip/hip_bf16.h>

// ---------------------------------------------------------------------------
// GQA attention for MI455X (gfx1250, wave32, WMMA 16x16x32 f16 -> f32 acc)
// Pipeline: f32->f16 converts, Q/K/V/O projections (WMMA GEMM with TDM
// tensor_load_to_lds double-buffered LDS staging), RoPE repack, flash
// attention (transposed S/O so softmax is lane-local), output GEMM.
// ---------------------------------------------------------------------------

typedef _Float16 v16h __attribute__((ext_vector_type(16)));
typedef _Float16 v8h  __attribute__((ext_vector_type(8)));
typedef float    v8f  __attribute__((ext_vector_type(8)));
typedef unsigned int uint32x4 __attribute__((ext_vector_type(4)));
typedef int          int32x4  __attribute__((ext_vector_type(4)));
typedef int          int32x8  __attribute__((ext_vector_type(8)));

#define TB   2
#define TT   2048
#define TC   2048
#define NH   32
#define NKVH 8
#define HD   64

union FragH  { v16h v; v8h h[2]; };
union FragHE { v16h v; _Float16 e[16]; };

#define WMMA_F16(A_, B_, C_) \
  __builtin_amdgcn_wmma_f32_16x16x32_f16(false, (A_), false, (B_), (short)0, (C_), false, false)

// ---------------------------------------------------------------------------
// TDM: 2-D tile DMA global -> LDS (CDNA5 ISA).
// data_size=1 (2-byte elements). tile_dim0 elements contiguous, tile_dim1 rows
// of stride0 elements. Tracked on TENSORcnt.
__device__ __forceinline__ void tdm_load_2d(unsigned lds_off, const void* gptr,
                                            unsigned tile_d0, unsigned tile_d1,
                                            unsigned tensor_d0, unsigned tensor_d1,
                                            unsigned long long stride0) {
  unsigned long long ga = (unsigned long long)(uintptr_t)gptr;
  uint32x4 g0;
  g0[0] = 1u;                                       // count=1, user D#
  g0[1] = lds_off;                                  // lds_addr (bytes)
  g0[2] = (unsigned)(ga & 0xffffffffull);           // global_addr[31:0]
  g0[3] = (unsigned)((ga >> 32) & 0x01ffffffull)    // global_addr[56:32]
        | (2u << 30);                               // type=2 ("image")
  int32x8 g1;
  g1[0] = (int)(1u << 16);                          // data_size=1 (2B); mask/pad/iter=0
  g1[1] = (int)((tensor_d0 & 0xffffu) << 16);       // [47:32]=barrier addr(0); [63:48]=dim0 lo
  g1[2] = (int)((tensor_d0 >> 16) | ((tensor_d1 & 0xffffu) << 16));
  g1[3] = (int)((tensor_d1 >> 16) | (tile_d0 << 16));
  g1[4] = (int)(tile_d1 & 0xffffu);                 // tile_dim1; tile_dim2=0
  g1[5] = (int)(unsigned)(stride0 & 0xffffffffull); // tensor_dim0_stride[31:0]
  g1[6] = (int)(unsigned)((stride0 >> 32) & 0xffffull); // stride[47:32]; dim1_stride lo=0
  g1[7] = 0;
  int32x4 z4 = {0, 0, 0, 0};
#if __clang_major__ >= 23
  int32x8 z8 = {0, 0, 0, 0, 0, 0, 0, 0};
  __builtin_amdgcn_tensor_load_to_lds(g0, g1, z4, z4, z8, 0);
#else
  __builtin_amdgcn_tensor_load_to_lds(g0, g1, z4, z4, 0);
#endif
}

// ---------------------------------------------------------------------------
__global__ void f32_to_f16_kernel(const float* __restrict__ src,
                                  _Float16* __restrict__ dst, size_t n) {
  size_t i = (size_t)blockIdx.x * blockDim.x + threadIdx.x;
  size_t stride = (size_t)gridDim.x * blockDim.x;
  for (; i < n; i += stride) dst[i] = (_Float16)src[i];
}

// out[M,N] = A[M,K] * W[N,K]^T + bias ; A,W f16, out f32.
// Block: 8 waves, 128x128 tile, K stepped by 32. A/B k-slices staged in LDS by
// the Tensor Data Mover (double-buffered), fragments read via ds_load_b128.
__launch_bounds__(256)
__global__ void gemm_xWT_bias(const _Float16* __restrict__ A,
                              const _Float16* __restrict__ W,
                              const float* __restrict__ bias,
                              float* __restrict__ out,
                              int M, int N, int K) {
  __shared__ __align__(16) _Float16 As[2][128 * 32];  // [row][k] row-major
  __shared__ __align__(16) _Float16 Bs[2][128 * 32];  // [ncol][k] row-major

  const int lane = threadIdx.x & 31;
  const int wave = threadIdx.x >> 5;
  const int wm = wave >> 2;          // 0..1 -> 64-row strip
  const int wn = wave & 3;           // 0..3 -> 32-col strip
  const int m0 = blockIdx.y * 128;
  const int n0 = blockIdx.x * 128;
  const int half = lane >> 4;
  const int m    = lane & 15;

  v8f acc[4][2];
#pragma unroll
  for (int i = 0; i < 4; ++i)
#pragma unroll
    for (int j = 0; j < 2; ++j)
      acc[i][j] = (v8f){0.f,0.f,0.f,0.f,0.f,0.f,0.f,0.f};

  auto issue = [&](int buf, int k) {
    tdm_load_2d((unsigned)(uintptr_t)&As[buf][0], A + (size_t)m0 * K + k,
                /*tile*/ 32u, 128u, /*tensor*/ (unsigned)K, (unsigned)M,
                (unsigned long long)K);
    tdm_load_2d((unsigned)(uintptr_t)&Bs[buf][0], W + (size_t)n0 * K + k,
                32u, 128u, (unsigned)K, (unsigned)N,
                (unsigned long long)K);
  };

  if (threadIdx.x < 32) issue(0, 0);  // prologue DMA (wave 0, once per wave)

  int buf = 0;
  for (int k = 0; k < K; k += 32) {
    if (threadIdx.x < 32) {
      if (k + 32 < K) {               // prefetch next slice into other buffer
        issue(buf ^ 1, k + 32);
        __builtin_amdgcn_s_wait_tensorcnt(2);  // in-order: current buffer done
      } else {
        __builtin_amdgcn_s_wait_tensorcnt(0);
      }
    }
    __syncthreads();                  // publish LDS tiles to all 8 waves

    const _Float16* at = &As[buf][0];
    const _Float16* bt = &Bs[buf][0];

    v16h bfr[2];
#pragma unroll
    for (int jn = 0; jn < 2; ++jn) {  // B frag: col = wn*32+jn*16+m, chunks 8h / 16+8h
      const _Float16* p = bt + (size_t)(wn * 32 + jn * 16 + m) * 32 + 8 * half;
      FragH f;
      f.h[0] = *(const v8h*)(p);
      f.h[1] = *(const v8h*)(p + 16);
      bfr[jn] = f.v;
    }
#pragma unroll
    for (int i = 0; i < 4; ++i) {     // A frag: row = wm*64+16i+m, chunks 8h / 16+8h
      const _Float16* p = at + (size_t)(wm * 64 + 16 * i + m) * 32 + 8 * half;
      FragH f;
      f.h[0] = *(const v8h*)(p);
      f.h[1] = *(const v8h*)(p + 16);
      acc[i][0] = WMMA_F16(f.v, bfr[0], acc[i][0]);
      acc[i][1] = WMMA_F16(f.v, bfr[1], acc[i][1]);
    }
    __syncthreads();                  // all reads done before TDM rewrites buf
    buf ^= 1;
  }

  // C/D layout: VGPR j -> row (+j+8*half); col = lane%16
#pragma unroll
  for (int i = 0; i < 4; ++i)
#pragma unroll
    for (int jn = 0; jn < 2; ++jn)
#pragma unroll
      for (int j = 0; j < 8; ++j) {
        int row = m0 + wm * 64 + 16 * i + j + 8 * half;
        int col = n0 + wn * 32 + 16 * jn + m;
        out[(size_t)row * N + col] = acc[i][jn][j] + bias[col];
      }
}

// RoPE + head-major repack: P f32 [B*T, nheads*64] -> out f16 [B,nheads,T,64].
__global__ void rope_pack_kernel(const float* __restrict__ P,
                                 const float* __restrict__ tab,
                                 _Float16* __restrict__ outp,
                                 int nheads, float scale) {
  size_t idx = (size_t)blockIdx.x * blockDim.x + threadIdx.x;
  size_t total = (size_t)TB * TT * nheads * 32;
  if (idx >= total) return;
  int d = (int)(idx & 31);
  size_t r = idx >> 5;
  int h = (int)(r % nheads); r /= nheads;
  int t = (int)(r % TT);
  int b = (int)(r / TT);
  const float* row = P + ((size_t)b * TT + t) * ((size_t)nheads * HD) + (size_t)h * HD;
  float q1 = row[d];
  float q2 = row[d + 32];
  const float* tp = tab + (size_t)t * HD;
  int e = 2 * (d >> 1);
  float c1 = tp[e],      s1 = tp[e + 1];
  float c2 = tp[e + 32], s2 = tp[e + 33];
  float o1 = (q1 * c1 - q2 * s1) * scale;
  float o2 = (q2 * c2 + q1 * s2) * scale;
  _Float16* dst = outp + (((size_t)b * nheads + h) * TT + t) * HD;
  dst[d]      = (_Float16)o1;
  dst[d + 32] = (_Float16)o2;
}

// V repack + transpose: P f32 [B*T, KVH*64] -> vt f16 [B,KVH,64,T]
__global__ void v_transpose_kernel(const float* __restrict__ P,
                                   _Float16* __restrict__ vt) {
  size_t idx = (size_t)blockIdx.x * blockDim.x + threadIdx.x;
  size_t total = (size_t)TB * TT * NKVH * HD;
  if (idx >= total) return;
  int d = (int)(idx & 63);
  size_t r = idx >> 6;
  int h = (int)(r % NKVH); r /= NKVH;
  int t = (int)(r % TT);
  int b = (int)(r / TT);
  float v = P[((size_t)b * TT + t) * ((size_t)NKVH * HD) + (size_t)h * HD + d];
  vt[(((size_t)b * NKVH + h) * HD + d) * TT + t] = (_Float16)v;
}

// Flash attention, transposed: S^T = K*Q^T, O^T = V^T*P^T.
// One wave per 16-query tile. Q pre-scaled by D^-0.5. Causal mask.
__launch_bounds__(128)
__global__ void flash_attn_kernel(const _Float16* __restrict__ Qh,
                                  const _Float16* __restrict__ Kh,
                                  const _Float16* __restrict__ Vt,
                                  _Float16* __restrict__ O) {
  const int lane = threadIdx.x & 31;
  const int wave = threadIdx.x >> 5;
  const int q0   = (blockIdx.x * 4 + wave) * 16;
  const int h    = blockIdx.y;
  const int b    = blockIdx.z;
  const int kvh  = h % NKVH;
  const int half = lane >> 4;
  const int c    = lane & 15;

  const _Float16* Qbase = Qh + ((size_t)(b * NH + h))     * TT * HD;
  const _Float16* Kbase = Kh + ((size_t)(b * NKVH + kvh)) * TT * HD;
  const _Float16* Vbase = Vt + ((size_t)(b * NKVH + kvh)) * HD * TT;

  v16h qf[2];
  {
    const _Float16* p = Qbase + (size_t)(q0 + c) * HD;
#pragma unroll
    for (int kk = 0; kk < 2; ++kk) {
      FragH f;
      const _Float16* pp = p + 32 * kk + 8 * half;
      f.h[0] = *(const v8h*)(pp);
      f.h[1] = *(const v8h*)(pp + 16);
      qf[kk] = f.v;
    }
  }

  v8f o[4];
#pragma unroll
  for (int mt = 0; mt < 4; ++mt) o[mt] = (v8f){0.f,0.f,0.f,0.f,0.f,0.f,0.f,0.f};
  float mstat = -1e30f, lstat = 0.f;
  const int qmax = q0 + 15;

  for (int k0 = 0; k0 <= qmax; k0 += 32) {
    v8f s[2];
#pragma unroll
    for (int tt = 0; tt < 2; ++tt) {
      s[tt] = (v8f){0.f,0.f,0.f,0.f,0.f,0.f,0.f,0.f};
#pragma unroll
      for (int kk = 0; kk < 2; ++kk) {
        const _Float16* p = Kbase + (size_t)(k0 + 16 * tt + c) * HD + 32 * kk + 8 * half;
        FragH f;
        f.h[0] = *(const v8h*)(p);
        f.h[1] = *(const v8h*)(p + 16);
        s[tt] = WMMA_F16(f.v, qf[kk], s[tt]);
      }
    }
#pragma unroll
    for (int tt = 0; tt < 2; ++tt)
#pragma unroll
      for (int j = 0; j < 8; ++j) {
        int key = k0 + 16 * tt + j + 8 * half;
        if (key > q0 + c) s[tt][j] = -1e30f;
      }
    float cm = -1e30f;
#pragma unroll
    for (int j = 0; j < 8; ++j) { cm = fmaxf(cm, s[0][j]); cm = fmaxf(cm, s[1][j]); }
    cm = fmaxf(cm, __shfl_xor(cm, 16, 32));
    float mnew  = fmaxf(mstat, cm);
    float alpha = __expf(mstat - mnew);
    float p0[8], p1[8];
    float rs = 0.f;
#pragma unroll
    for (int j = 0; j < 8; ++j) {
      p0[j] = __expf(s[0][j] - mnew);
      p1[j] = __expf(s[1][j] - mnew);
      rs += p0[j] + p1[j];
    }
    rs += __shfl_xor(rs, 16, 32);
    lstat = lstat * alpha + rs;
    mstat = mnew;
#pragma unroll
    for (int mt = 0; mt < 4; ++mt)
#pragma unroll
      for (int j = 0; j < 8; ++j) o[mt][j] *= alpha;

    FragHE pf;  // C-layout -> B-fragment via one half-swap shuffle per element
#pragma unroll
    for (int j = 0; j < 8; ++j) {
      float a0 = p0[j], a1 = p1[j];
      float x0 = __shfl_xor(a0, 16, 32);
      float x1 = __shfl_xor(a1, 16, 32);
      if (half == 0) { pf.e[j] = (_Float16)a0; pf.e[8 + j] = (_Float16)x0; }
      else           { pf.e[j] = (_Float16)x1; pf.e[8 + j] = (_Float16)a1; }
    }
#pragma unroll
    for (int mt = 0; mt < 4; ++mt) {
      const _Float16* p = Vbase + (size_t)(16 * mt + c) * TT + k0 + 16 * half;
      FragH f;
      f.h[0] = *(const v8h*)(p);
      f.h[1] = *(const v8h*)(p + 8);
      o[mt] = WMMA_F16(f.v, pf.v, o[mt]);
    }
  }

  const float inv = 1.f / lstat;
  const int t = q0 + c;
  _Float16* dst = O + ((size_t)b * TT + t) * TC + (size_t)h * HD;
#pragma unroll
  for (int mt = 0; mt < 4; ++mt)
#pragma unroll
    for (int j = 0; j < 8; ++j)
      dst[16 * mt + j + 8 * half] = (_Float16)(o[mt][j] * inv);
}

// ---------------------------------------------------------------------------
extern "C" void kernel_launch(void* const* d_in, const int* in_sizes, int n_in,
                              void* d_out, int out_size, void* d_ws, size_t ws_size,
                              hipStream_t stream) {
  (void)in_sizes; (void)n_in; (void)out_size; (void)ws_size;
  const float* x   = (const float*)d_in[0];
  const float* tab = (const float*)d_in[2];
  const float* Wq  = (const float*)d_in[3];
  const float* bq  = (const float*)d_in[4];
  const float* Wk  = (const float*)d_in[5];
  const float* bk  = (const float*)d_in[6];
  const float* Wv  = (const float*)d_in[7];
  const float* bv  = (const float*)d_in[8];
  const float* Wo  = (const float*)d_in[9];
  const float* bo  = (const float*)d_in[10];
  float* out = (float*)d_out;

  char* ws = (char*)d_ws;
  size_t off = 0;
  auto alloc = [&](size_t bytes) -> void* {
    void* p = ws + off;
    off = (off + bytes + 255) & ~(size_t)255;
    return p;
  };

  const size_t MT  = (size_t)TB * TT;   // 4096
  const size_t CKV = (size_t)NKVH * HD; // 512

  _Float16* xh   = (_Float16*)alloc(MT * TC * 2);
  _Float16* wqh  = (_Float16*)alloc((size_t)TC * TC * 2);
  _Float16* wkh  = (_Float16*)alloc(CKV * TC * 2);
  _Float16* wvh  = (_Float16*)alloc(CKV * TC * 2);
  _Float16* woh  = (_Float16*)alloc((size_t)TC * TC * 2);
  float*    qf   = (float*)   alloc(MT * TC * 4);
  float*    kf   = (float*)   alloc(MT * CKV * 4);
  float*    vf   = (float*)   alloc(MT * CKV * 4);
  _Float16* qh   = (_Float16*)alloc(MT * TC * 2);
  _Float16* kh   = (_Float16*)alloc(MT * CKV * 2);
  _Float16* vt   = (_Float16*)alloc(MT * CKV * 2);
  _Float16* attn = (_Float16*)qf;  // alias: qf (32MB) dead after rope

  auto cvt = [&](const float* s, _Float16* d, size_t n) {
    unsigned blocks = (unsigned)((n + 255) / 256);
    f32_to_f16_kernel<<<blocks, 256, 0, stream>>>(s, d, n);
  };
  cvt(x,  xh,  MT * TC);
  cvt(Wq, wqh, (size_t)TC * TC);
  cvt(Wk, wkh, CKV * TC);
  cvt(Wv, wvh, CKV * TC);
  cvt(Wo, woh, (size_t)TC * TC);

  gemm_xWT_bias<<<dim3(TC / 128, (unsigned)(MT / 128)), 256, 0, stream>>>(
      xh, wqh, bq, qf, (int)MT, TC, TC);
  gemm_xWT_bias<<<dim3((unsigned)(CKV / 128), (unsigned)(MT / 128)), 256, 0, stream>>>(
      xh, wkh, bk, kf, (int)MT, (int)CKV, TC);
  gemm_xWT_bias<<<dim3((unsigned)(CKV / 128), (unsigned)(MT / 128)), 256, 0, stream>>>(
      xh, wvh, bv, vf, (int)MT, (int)CKV, TC);

  {
    size_t nq = (size_t)TB * TT * NH * 32;
    rope_pack_kernel<<<(unsigned)((nq + 255) / 256), 256, 0, stream>>>(qf, tab, qh, NH, 0.125f);
    size_t nk = (size_t)TB * TT * NKVH * 32;
    rope_pack_kernel<<<(unsigned)((nk + 255) / 256), 256, 0, stream>>>(kf, tab, kh, NKVH, 1.0f);
    size_t nv = (size_t)TB * TT * NKVH * HD;
    v_transpose_kernel<<<(unsigned)((nv + 255) / 256), 256, 0, stream>>>(vf, vt);
  }

  flash_attn_kernel<<<dim3(TT / 64, NH, TB), 128, 0, stream>>>(qh, kh, vt, attn);

  gemm_xWT_bias<<<dim3(TC / 128, (unsigned)(MT / 128)), 256, 0, stream>>>(
      attn, woh, bo, out, (int)MT, TC, TC);
}
